// SoftMoE_22720376996420
// MI455X (gfx1250) — matrix operand order repaired and checked by reading, not compiled, not run on hardware
//
#include <hip/hip_runtime.h>
#include <hip/hip_bf16.h>
#include <stdint.h>

// ---------------- problem constants ----------------
#define B_  4
#define L_  2048
#define D_  1024
#define H_  16
#define S_  512
#define E_  8
#define FF_ 4096
#define DH_ 64
#define EPS_ 1e-5f

// ---------------- CDNA5 WMMA types ----------------
typedef __attribute__((ext_vector_type(16))) __bf16 v16bf;
typedef __attribute__((ext_vector_type(8)))  float  v8f;
typedef __attribute__((ext_vector_type(4)))  int    v4i;

__device__ __forceinline__ unsigned short f2bf(float f) {
  unsigned int u = __float_as_uint(f);
  u += 0x7fffu + ((u >> 16) & 1u);   // round-to-nearest-even
  return (unsigned short)(u >> 16);
}

union Frag {
  uint4 q[2];
  v16bf v;
};

// ---------------- CDNA5 async global->LDS copy (ASYNCcnt path) ----------------
#if __has_builtin(__builtin_amdgcn_global_load_async_to_lds_b128)
#define HAVE_ASYNC_LDS 1
__device__ __forceinline__ void async_copy_b128(const unsigned short* g, unsigned short* l) {
  // prototype (from hipcc diagnostic): (v4i32* global_src, v4i32* lds_dst, imm offset, imm cpol)
  __builtin_amdgcn_global_load_async_to_lds_b128((v4i*)(void*)g, (v4i*)(void*)l, 0, 0);
}
__device__ __forceinline__ void wait_async() {
#if __has_builtin(__builtin_amdgcn_s_wait_asynccnt)
  __builtin_amdgcn_s_wait_asynccnt(0);
#else
  asm volatile("s_wait_asynccnt 0x0" ::: "memory");
#endif
}
#else
#define HAVE_ASYNC_LDS 0
#endif

// ---------------- generic batched GEMM-NT: C = A(MxK) * B(NxK)^T ----------------
struct GemmParams {
  const unsigned short* A;
  const unsigned short* Bm;
  void* C;
  const float* bias;
  long strideA, strideB, strideC;
  long lda, ldb, ldc;
  int  aMod, bMod, biasMod;
  long biasStride;
  int  M, N, K;
  int  mode;              // see storeC
  int  pL, pH, pDH, pD;   // layout params for split/merge stores
};

#define BK 32
#define LDP 40   // row pitch in halves: 80B, keeps 16B alignment, skews banks

__device__ __forceinline__ void storeC(const GemmParams& p, int z, int m, int n, float v) {
  switch (p.mode) {
    case 0: {  // plain f32, batched
      float* C = (float*)p.C + (long)z * p.strideC;
      C[(long)m * p.ldc + n] = v;
    } break;
    case 1: {  // plain bf16, batched
      unsigned short* C = (unsigned short*)p.C + (long)z * p.strideC;
      C[(long)m * p.ldc + n] = f2bf(v);
    } break;
    case 2: {  // bias + relu^2 -> bf16 (expert up-proj)
      const float* bias = p.bias + (long)(z % p.biasMod) * p.biasStride;
      float t = v + bias[n];
      t = t > 0.f ? t * t : 0.f;
      unsigned short* C = (unsigned short*)p.C + (long)z * p.strideC;
      C[(long)m * p.ldc + n] = f2bf(t);
    } break;
    case 3: {  // bias -> f32 (expert down-proj)
      const float* bias = p.bias + (long)(z % p.biasMod) * p.biasStride;
      float* C = (float*)p.C + (long)z * p.strideC;
      C[(long)m * p.ldc + n] = v + bias[n];
    } break;
    case 4: {  // projection split-head: m=(b,l), n=(h,d) -> (B,H,L,DH) bf16
      int b = m / p.pL, l = m % p.pL, h = n / p.pDH, d = n % p.pDH;
      ((unsigned short*)p.C)[(((long)(b * p.pH + h) * p.pL + l) * p.pDH) + d] = f2bf(v);
    } break;
    case 5: {  // projection split-head transposed -> (B,H,DH,L) bf16
      int b = m / p.pL, l = m % p.pL, h = n / p.pDH, d = n % p.pDH;
      ((unsigned short*)p.C)[(((long)(b * p.pH + h) * p.pDH + d) * p.pL) + l] = f2bf(v);
    } break;
    case 6: {  // merge heads -> f32 (b = z/H, h = z%H), rows-per-batch = M
      int b = z / p.pH, h = z % p.pH;
      ((float*)p.C)[((long)b * p.M + m) * p.pD + h * p.pDH + n] = v;
    } break;
    case 7: {  // merge heads -> bf16
      int b = z / p.pH, h = z % p.pH;
      ((unsigned short*)p.C)[((long)b * p.M + m) * p.pD + h * p.pDH + n] = f2bf(v);
    } break;
  }
}

// BMT x BNT block tile, 8 waves arranged WR x WC, double-buffered LDS,
// async global->LDS pipeline when available.
template <int BMT, int BNT, int WR, int WC>
__global__ __launch_bounds__(256) void wmma_gemm_nt(GemmParams p) {
  constexpr int MT = BMT / (WR * 16);   // M subtiles per wave
  constexpr int NT = BNT / (WC * 16);   // N subtiles per wave

  __shared__ unsigned short As[2][BMT][LDP];
  __shared__ unsigned short Bs[2][BNT][LDP];

  const int z = blockIdx.z;
  const unsigned short* Ab = p.A  + (long)(z % p.aMod) * p.strideA;
  const unsigned short* Bb = p.Bm + (long)(z % p.bMod) * p.strideB;

  const int tileM = blockIdx.y * BMT;
  const int tileN = blockIdx.x * BNT;
  const int tid  = threadIdx.x;
  const int lane = tid & 31;
  const int wave = tid >> 5;
  const int wm = (wave / WC) * (MT * 16);
  const int wn = (wave % WC) * (NT * 16);

  v8f zero = {};
  v8f acc[MT][NT];
#pragma unroll
  for (int i = 0; i < MT; ++i)
#pragma unroll
    for (int j = 0; j < NT; ++j) acc[i][j] = zero;

  // cooperative tile loaders
  constexpr int ATH = 256 / BMT;        // threads per A row
  constexpr int AH  = 32 / ATH;         // halves per thread (16 or 8)
  constexpr int BTH = 256 / BNT;
  constexpr int BH  = 32 / BTH;
  const int arow = tid / ATH;
  const int acol = (tid % ATH) * AH;
  const int brow = tid / BTH;
  const int bcol = (tid % BTH) * BH;
  const int gmA = tileM + arow;
  const int gnB = tileN + brow;
  const bool aIn = gmA < p.M;
  const bool bIn = gnB < p.N;
  const unsigned short* aPtr = Ab + (long)gmA * p.lda + acol;
  const unsigned short* bPtr = Bb + (long)gnB * p.ldb + bcol;

  // zero OOB rows once (both buffers); loads never touch them again
  {
    uint4 zz = {0, 0, 0, 0};
    if (!aIn) {
#pragma unroll
      for (int c = 0; c < AH / 8; ++c) {
        *(uint4*)&As[0][arow][acol + c * 8] = zz;
        *(uint4*)&As[1][arow][acol + c * 8] = zz;
      }
    }
    if (!bIn) {
#pragma unroll
      for (int c = 0; c < BH / 8; ++c) {
        *(uint4*)&Bs[0][brow][bcol + c * 8] = zz;
        *(uint4*)&Bs[1][brow][bcol + c * 8] = zz;
      }
    }
  }

  const int nk = p.K / BK;

  auto loadTile = [&](int kt, int buf) {
    const int k0 = kt * BK;
#if HAVE_ASYNC_LDS
    if (aIn) {
#pragma unroll
      for (int c = 0; c < AH / 8; ++c)
        async_copy_b128(aPtr + k0 + c * 8, &As[buf][arow][acol + c * 8]);
    }
    if (bIn) {
#pragma unroll
      for (int c = 0; c < BH / 8; ++c)
        async_copy_b128(bPtr + k0 + c * 8, &Bs[buf][brow][bcol + c * 8]);
    }
#else
    if (aIn) {
#pragma unroll
      for (int c = 0; c < AH / 8; ++c)
        *(uint4*)&As[buf][arow][acol + c * 8] = *(const uint4*)(aPtr + k0 + c * 8);
    }
    if (bIn) {
#pragma unroll
      for (int c = 0; c < BH / 8; ++c)
        *(uint4*)&Bs[buf][brow][bcol + c * 8] = *(const uint4*)(bPtr + k0 + c * 8);
    }
    if (kt + 1 < nk) {
      if (aIn) __builtin_prefetch(aPtr + k0 + BK, 0, 1);
      if (bIn) __builtin_prefetch(bPtr + k0 + BK, 0, 1);
    }
#endif
  };

  auto syncLoad = [&]() {
#if HAVE_ASYNC_LDS
    wait_async();
#endif
    __syncthreads();
  };

  // fragment index math (CDNA5 16-bit WMMA VGPR layouts)
  const int fm  = lane & 15;
  const int fko = (lane >> 4) * 8;    // A: lanes0-15 K={0..7,16..23}; lanes16-31 K={8..15,24..31}
  const int fkh = (lane >> 4) * 8;    // B: lanes0-15 K={0..7,16..23}; lanes16-31 K={8..15,24..31}

  auto compute = [&](int buf) {
    Frag af[MT], bfr[NT];
#pragma unroll
    for (int i = 0; i < MT; ++i) {
      const unsigned short* r = &As[buf][wm + i * 16 + fm][0];
      af[i].q[0] = *(const uint4*)(r + fko);
      af[i].q[1] = *(const uint4*)(r + fko + 16);
    }
#pragma unroll
    for (int j = 0; j < NT; ++j) {
      const unsigned short* r = &Bs[buf][wn + j * 16 + fm][0];
      bfr[j].q[0] = *(const uint4*)(r + fkh);
      bfr[j].q[1] = *(const uint4*)(r + fkh + 16);
    }
#pragma unroll
    for (int i = 0; i < MT; ++i)
#pragma unroll
      for (int j = 0; j < NT; ++j)
        acc[i][j] = __builtin_amdgcn_wmma_f32_16x16x32_bf16(
            false, af[i].v, false, bfr[j].v, (short)0, acc[i][j], false, false);
  };

  // double-buffered pipeline: load(k+1) overlaps compute(k)
  loadTile(0, 0);
  syncLoad();
  for (int kt = 0; kt < nk; ++kt) {
    const int cur = kt & 1;
    if (kt + 1 < nk) loadTile(kt + 1, cur ^ 1);
    compute(cur);
    syncLoad();
  }

  // epilogue: C 16x16 f32: VGPR r -> M=r (lanes0-15) / M=r+8 (lanes16-31), N=lane&15
  const int colL = lane & 15;
  const int rowH = (lane >> 4) * 8;
#pragma unroll
  for (int i = 0; i < MT; ++i)
#pragma unroll
    for (int j = 0; j < NT; ++j)
#pragma unroll
      for (int r = 0; r < 8; ++r) {
        int m = tileM + wm + i * 16 + rowH + r;
        int n = tileN + wn + j * 16 + colL;
        if (m < p.M && n < p.N) storeC(p, z, m, n, acc[i][j][r]);
      }
}

// ---------------- block reductions (wave32) ----------------
__device__ __forceinline__ float blockReduceSum(float v, float* red) {
#pragma unroll
  for (int off = 16; off > 0; off >>= 1) v += __shfl_xor(v, off, 32);
  int lane = threadIdx.x & 31, wave = threadIdx.x >> 5;
  if (lane == 0) red[wave] = v;
  __syncthreads();
  float t = (threadIdx.x < 8) ? red[threadIdx.x] : 0.f;
  if (wave == 0) {
#pragma unroll
    for (int off = 4; off > 0; off >>= 1) t += __shfl_xor(t, off, 32);
    if (lane == 0) red[0] = t;
  }
  __syncthreads();
  float r = red[0];
  __syncthreads();
  return r;
}

__device__ __forceinline__ float blockReduceMax(float v, float* red) {
#pragma unroll
  for (int off = 16; off > 0; off >>= 1) v = fmaxf(v, __shfl_xor(v, off, 32));
  int lane = threadIdx.x & 31, wave = threadIdx.x >> 5;
  if (lane == 0) red[wave] = v;
  __syncthreads();
  float t = (threadIdx.x < 8) ? red[threadIdx.x] : -3.4e38f;
  if (wave == 0) {
#pragma unroll
    for (int off = 4; off > 0; off >>= 1) t = fmaxf(t, __shfl_xor(t, off, 32));
    if (lane == 0) red[0] = t;
  }
  __syncthreads();
  float r = red[0];
  __syncthreads();
  return r;
}

// ---------------- LayerNorm -> bf16 ----------------
__global__ __launch_bounds__(256) void layernorm_bf16(
    const float* __restrict__ x, const float* __restrict__ w,
    const float* __restrict__ b, unsigned short* __restrict__ out, int D) {
  __shared__ float red[8];
  long row = blockIdx.x;
  const float* xr = x + row * (long)D;
  float s = 0.f, s2 = 0.f;
  for (int i = threadIdx.x; i < D; i += 256) { float v = xr[i]; s += v; s2 += v * v; }
  s  = blockReduceSum(s, red);
  s2 = blockReduceSum(s2, red);
  float mean = s / (float)D;
  float var  = s2 / (float)D - mean * mean;
  float inv  = rsqrtf(var + EPS_);
  unsigned short* o = out + row * (long)D;
  for (int i = threadIdx.x; i < D; i += 256)
    o[i] = f2bf((xr[i] - mean) * inv * w[i] + b[i]);
}

// ---------------- scaled softmax over rows, in-place f32 -> bf16 ----------------
__global__ __launch_bounds__(256) void softmax_bf16_inplace(
    float* __restrict__ buf, int rowLen, float scale) {
  __shared__ float red[8];
  long row = blockIdx.x;
  float* r = buf + row * (long)rowLen;
  unsigned short* out = (unsigned short*)r;
  float v[8];
  int cnt = rowLen >> 8;
  float mx = -3.4e38f;
  for (int c = 0; c < cnt; ++c) {
    v[c] = r[c * 256 + threadIdx.x] * scale;
    mx = fmaxf(mx, v[c]);
  }
  mx = blockReduceMax(mx, red);
  float s = 0.f;
  for (int c = 0; c < cnt; ++c) { v[c] = __expf(v[c] - mx); s += v[c]; }
  s = blockReduceSum(s, red);     // barrier inside: all reads done before writes
  float inv = 1.f / s;
  for (int c = 0; c < cnt; ++c) out[c * 256 + threadIdx.x] = f2bf(v[c] * inv);
}

// ---------------- f32 -> bf16 convert ----------------
__global__ __launch_bounds__(256) void cvt_f32_bf16(
    const float* __restrict__ in, unsigned short* __restrict__ out, long n) {
  long i = (long)blockIdx.x * 256 + threadIdx.x;
  if (i < n) out[i] = f2bf(in[i]);
}

// ---------------- eo (B,S,D) f32 -> vcT (B,H,DH,S) bf16 ----------------
__global__ __launch_bounds__(256) void make_vcT_kernel(
    const float* __restrict__ eo, unsigned short* __restrict__ vcT) {
  long i = (long)blockIdx.x * 256 + threadIdx.x;
  const long total = (long)B_ * H_ * DH_ * S_;
  if (i >= total) return;
  int s = (int)(i % S_); long t = i / S_;
  int d = (int)(t % DH_); t /= DH_;
  int h = (int)(t % H_);
  int b = (int)(t / H_);
  vcT[i] = f2bf(eo[((long)(b * S_ + s)) * D_ + h * DH_ + d]);
}

// ---------------- host orchestration ----------------
// cfg 0: 128x128 tiles (2x4 waves, 4x2 subtiles) - big GEMMs
// cfg 1: 128x64  tiles (4x2 waves, 2x2 subtiles) - N=64 GEMMs
// cfg 2: 64x128  tiles (2x4 waves, 2x2 subtiles) - M=64 GEMMs
static void launch_gemm(hipStream_t st, const GemmParams& p, int batches, int cfg) {
  if (cfg == 0) {
    dim3 grid((p.N + 127) / 128, (p.M + 127) / 128, batches);
    wmma_gemm_nt<128, 128, 2, 4><<<grid, dim3(256), 0, st>>>(p);
  } else if (cfg == 1) {
    dim3 grid((p.N + 63) / 64, (p.M + 127) / 128, batches);
    wmma_gemm_nt<128, 64, 4, 2><<<grid, dim3(256), 0, st>>>(p);
  } else {
    dim3 grid((p.N + 127) / 128, (p.M + 63) / 64, batches);
    wmma_gemm_nt<64, 128, 2, 4><<<grid, dim3(256), 0, st>>>(p);
  }
}

extern "C" void kernel_launch(void* const* d_in, const int* in_sizes, int n_in,
                              void* d_out, int out_size, void* d_ws, size_t ws_size,
                              hipStream_t stream) {
  (void)in_sizes; (void)n_in; (void)out_size; (void)ws_size;

  const float* x     = (const float*)d_in[0];
  const float* sq_f  = (const float*)d_in[1];
  const float* sk_f  = (const float*)d_in[2];
  const float* wk_f  = (const float*)d_in[3];
  const float* wv_f  = (const float*)d_in[4];
  const float* wq_f  = (const float*)d_in[5];
  const float* win_f = (const float*)d_in[6];
  const float* bin   = (const float*)d_in[7];
  const float* wout_f= (const float*)d_in[8];
  const float* bout  = (const float*)d_in[9];
  const float* pnw   = (const float*)d_in[10];
  const float* pnb   = (const float*)d_in[11];

  char* ws = (char*)d_ws;
  size_t off = 0;
  auto alloc = [&](size_t bytes) -> void* {
    void* p = ws + off;
    off += (bytes + 255) & ~(size_t)255;
    return p;
  };

  unsigned short* xn   = (unsigned short*)alloc((size_t)B_ * L_ * D_ * 2);
  unsigned short* wk   = (unsigned short*)alloc((size_t)D_ * D_ * 2);
  unsigned short* wv   = (unsigned short*)alloc((size_t)D_ * D_ * 2);
  unsigned short* wq   = (unsigned short*)alloc((size_t)D_ * D_ * 2);
  unsigned short* sq   = (unsigned short*)alloc((size_t)H_ * S_ * DH_ * 2);
  unsigned short* sk   = (unsigned short*)alloc((size_t)H_ * S_ * DH_ * 2);
  unsigned short* kbuf = (unsigned short*)alloc((size_t)B_ * L_ * D_ * 2);   // (B,H,L,DH)
  unsigned short* vT   = (unsigned short*)alloc((size_t)B_ * L_ * D_ * 2);   // (B,H,DH,L)
  unsigned short* qbuf = (unsigned short*)alloc((size_t)B_ * L_ * D_ * 2);   // (B,H,L,DH)
  unsigned short* ybuf = (unsigned short*)alloc((size_t)B_ * S_ * D_ * 2);
  unsigned short* win  = (unsigned short*)alloc((size_t)E_ * FF_ * D_ * 2);
  unsigned short* wout = (unsigned short*)alloc((size_t)E_ * D_ * FF_ * 2);
  unsigned short* hbuf = (unsigned short*)alloc((size_t)B_ * S_ * FF_ * 2);
  float*          eo   = (float*)alloc((size_t)B_ * S_ * D_ * 4);
  unsigned short* vcT  = (unsigned short*)alloc((size_t)B_ * H_ * DH_ * S_ * 2);
  float*          sc   = (float*)alloc((size_t)B_ * H_ * S_ * L_ * 4);       // reused

  auto cvt = [&](const float* in, unsigned short* out, long n) {
    cvt_f32_bf16<<<dim3((unsigned)((n + 255) / 256)), dim3(256), 0, stream>>>(in, out, n);
  };

  // 1) weights/constants -> bf16
  cvt(wk_f,  wk,  (long)D_ * D_);
  cvt(wv_f,  wv,  (long)D_ * D_);
  cvt(wq_f,  wq,  (long)D_ * D_);
  cvt(sq_f,  sq,  (long)H_ * S_ * DH_);
  cvt(sk_f,  sk,  (long)H_ * S_ * DH_);
  cvt(win_f, win, (long)E_ * FF_ * D_);
  cvt(wout_f,wout,(long)E_ * D_ * FF_);

  // 2) layernorm
  layernorm_bf16<<<dim3(B_ * L_), dim3(256), 0, stream>>>(x, pnw, pnb, xn, D_);

  GemmParams g;

  // 3) k = xn @ Wk^T  -> split heads (B,H,L,DH)
  g = GemmParams{};
  g.A = xn; g.aMod = 1; g.strideA = 0; g.lda = D_;
  g.Bm = wk; g.bMod = 1; g.strideB = 0; g.ldb = D_;
  g.C = kbuf; g.M = B_ * L_; g.N = D_; g.K = D_;
  g.mode = 4; g.pL = L_; g.pH = H_; g.pDH = DH_; g.pD = D_;
  launch_gemm(stream, g, 1, 0);

  // 4) v = xn @ Wv^T -> split heads transposed (B,H,DH,L)
  g.Bm = wv; g.C = vT; g.mode = 5;
  launch_gemm(stream, g, 1, 0);

  // 5) q = xn @ Wq^T -> split heads (B,H,L,DH)
  g.Bm = wq; g.C = qbuf; g.mode = 4;
  launch_gemm(stream, g, 1, 0);

  // 6) dispatch scores[z=(b,h)] = slot_q[h] (SxDH) @ k[b,h]^T (LxDH) -> (S,L) f32
  g = GemmParams{};
  g.A = sq;    g.aMod = H_;      g.strideA = (long)S_ * DH_; g.lda = DH_;
  g.Bm = kbuf; g.bMod = B_ * H_; g.strideB = (long)L_ * DH_; g.ldb = DH_;
  g.C = sc; g.strideC = (long)S_ * L_; g.ldc = L_;
  g.M = S_; g.N = L_; g.K = DH_; g.mode = 0;
  launch_gemm(stream, g, B_ * H_, 0);

  // 7) softmax over L, scale 1/sqrt(64), in-place -> bf16
  softmax_bf16_inplace<<<dim3(B_ * H_ * S_), dim3(256), 0, stream>>>(sc, L_, 0.125f);

  // 8) y[z] = p (SxL) @ vT[z]^T (DHxL) -> merged heads bf16 (B,S,D)
  g = GemmParams{};
  g.A = (const unsigned short*)sc; g.aMod = B_ * H_; g.strideA = (long)S_ * 2 * L_; g.lda = 2 * L_;
  g.Bm = vT; g.bMod = B_ * H_; g.strideB = (long)DH_ * L_; g.ldb = L_;
  g.C = ybuf; g.M = S_; g.N = DH_; g.K = L_;
  g.mode = 7; g.pH = H_; g.pDH = DH_; g.pD = D_;
  launch_gemm(stream, g, B_ * H_, 1);

  // 9) experts up: h[z=(b,e)] = y_be (64xD) @ w_in[e]^T (FFxD), bias+relu^2 -> bf16
  g = GemmParams{};
  g.A = ybuf; g.aMod = B_ * E_; g.strideA = (long)(S_ / E_) * D_; g.lda = D_;
  g.Bm = win; g.bMod = E_; g.strideB = (long)FF_ * D_; g.ldb = D_;
  g.C = hbuf; g.strideC = (long)(S_ / E_) * FF_; g.ldc = FF_;
  g.bias = bin; g.biasMod = E_; g.biasStride = FF_;
  g.M = S_ / E_; g.N = FF_; g.K = D_; g.mode = 2;
  launch_gemm(stream, g, B_ * E_, 2);

  // 10) experts down: eo[z] = h (64xFF) @ w_out[e]^T (DxFF) + b_out -> f32
  g = GemmParams{};
  g.A = hbuf; g.aMod = B_ * E_; g.strideA = (long)(S_ / E_) * FF_; g.lda = FF_;
  g.Bm = wout; g.bMod = E_; g.strideB = (long)D_ * FF_; g.ldb = FF_;
  g.C = eo; g.strideC = (long)(S_ / E_) * D_; g.ldc = D_;
  g.bias = bout; g.biasMod = E_; g.biasStride = D_;
  g.M = S_ / E_; g.N = D_; g.K = FF_; g.mode = 3;
  launch_gemm(stream, g, B_ * E_, 2);

  // 11) vcT (B,H,DH,S) bf16 from eo
  {
    long total = (long)B_ * H_ * DH_ * S_;
    make_vcT_kernel<<<dim3((unsigned)((total + 255) / 256)), dim3(256), 0, stream>>>(eo, vcT);
  }

  // 12) combine scores[z=(b,h)] = q[b,h] (LxDH) @ slot_key[h]^T (SxDH) -> (L,S) f32
  g = GemmParams{};
  g.A = qbuf; g.aMod = B_ * H_; g.strideA = (long)L_ * DH_; g.lda = DH_;
  g.Bm = sk; g.bMod = H_; g.strideB = (long)S_ * DH_; g.ldb = DH_;
  g.C = sc; g.strideC = (long)L_ * S_; g.ldc = S_;
  g.M = L_; g.N = S_; g.K = DH_; g.mode = 0;
  launch_gemm(stream, g, B_ * H_, 0);

  // 13) softmax over S, in-place -> bf16
  softmax_bf16_inplace<<<dim3(B_ * H_ * L_), dim3(256), 0, stream>>>(sc, S_, 0.125f);

  // 14) o[z] = p (LxS) @ vcT[z]^T (DHxS) -> merged heads f32 into d_out (B,L,D)
  g = GemmParams{};
  g.A = (const unsigned short*)sc; g.aMod = B_ * H_; g.strideA = (long)L_ * 2 * S_; g.lda = 2 * S_;
  g.Bm = vcT; g.bMod = B_ * H_; g.strideB = (long)DH_ * S_; g.ldb = S_;
  g.C = d_out; g.M = L_; g.N = DH_; g.K = S_;
  g.mode = 6; g.pH = H_; g.pDH = DH_; g.pD = D_;
  launch_gemm(stream, g, B_ * H_, 1);
}
